// SimpleLTM_61804579389943
// MI455X (gfx1250) — matrix operand order repaired and checked by reading, not compiled, not run on hardware
//
#include <hip/hip_runtime.h>
#include <hip/hip_bf16.h>
#include <float.h>

typedef __attribute__((ext_vector_type(16))) __bf16 bf16x16;
typedef __attribute__((ext_vector_type(8)))  __bf16 bf16x8;
typedef __attribute__((ext_vector_type(8)))  float  floatx8;

struct Cand { float v; int i; };

#define D 512
#define KSTEPS 16          // 512 / 32
#define GRP 4              // K-steps per B register group (double buffered)
#define QTILE 32           // queries per workgroup (two 16-row subtiles)
#define NWAVES 8
#define NSPLITS 8
#define LDSROW 520         // Q tile LDS row stride in bf16 (1040B: bank-spread)
#define SQ_BYTES (QTILE * LDSROW * 2)        // 33280
#define SC_BYTES (NWAVES * 2 * 16 * 20 * 4)  // 20480: per-wave, per-subtile C scratch

#define SHUF16(a, b) __builtin_shufflevector(a, b, 0, 1, 2, 3, 4, 5, 6, 7, \
                                             8, 9, 10, 11, 12, 13, 14, 15)

// Branchless sorted-descending top-8 insertion (guarded by current min).
#define TOP8_INSERT(cv_, ci_, TV, TI)                          \
  if ((cv_) > (TV)[7]) {                                       \
    float _c = (cv_); int _i = (ci_);                          \
    _Pragma("unroll")                                          \
    for (int _j = 0; _j < 8; ++_j) {                           \
      bool _g = _c > (TV)[_j];                                 \
      float _nv = _g ? _c : (TV)[_j];                          \
      float _nc = _g ? (TV)[_j] : _c;                          \
      int   _ni = _g ? _i : (TI)[_j];                          \
      int   _nx = _g ? (TI)[_j] : _i;                          \
      (TV)[_j] = _nv; (TI)[_j] = _ni; _c = _nc; _i = _nx;      \
    }                                                          \
  }

// ---------------- row L2-normalize: f32 -> bf16 ----------------
__global__ __launch_bounds__(256) void ltm_normalize_bf16(
    const float* __restrict__ in, __bf16* __restrict__ out, int rows) {
  const int lane = threadIdx.x & 31;
  const int wave = threadIdx.x >> 5;
  const int row  = blockIdx.x * NWAVES + wave;
  if (row >= rows) return;
  const float* r = in + (size_t)row * D;
  float s = 0.f;
#pragma unroll
  for (int k = 0; k < D / 32; ++k) { float x = r[lane + 32 * k]; s += x * x; }
#pragma unroll
  for (int off = 16; off >= 1; off >>= 1) s += __shfl_xor(s, off, 32);
  const float scale = 1.0f / fmaxf(sqrtf(s), 1e-12f);
  __bf16* o = out + (size_t)row * D;
#pragma unroll
  for (int k = 0; k < D / 32; ++k)
    o[lane + 32 * k] = (__bf16)(r[lane + 32 * k] * scale);
}

// ------- fused bf16-WMMA GEMM (32q x split-of-keys) + per-row top-8 -------
__global__ __launch_bounds__(256) void ltm_topk_partial(
    const __bf16* __restrict__ qn, const __bf16* __restrict__ kn,
    Cand* __restrict__ cand, int N, int splitLen) {
  __shared__ __align__(16) char smem[SQ_BYTES + SC_BYTES];
  __bf16* sQ = (__bf16*)smem;                   // Q tile (main loop)
  float*  sC = (float*)(smem + SQ_BYTES);       // per-wave/subtile C scratch
  Cand*   sLists = (Cand*)smem;                 // overlays sQ after main loop

  const int lane  = threadIdx.x & 31;
  const int wave  = threadIdx.x >> 5;
  const int row16 = lane & 15;
  const int hi    = lane >> 4;
  const int qtile = blockIdx.x;
  const int split = blockIdx.y;

  // ---- cooperative load of 32x512 Q tile into LDS (padded rows) ----
  {
    const int r = threadIdx.x >> 3;     // 0..31
    const int c = threadIdx.x & 7;      // 0..7, each covers 64 bf16
    const __bf16* src = qn + (size_t)(qtile * QTILE + r) * D + c * 64;
    __bf16* dst = sQ + r * LDSROW + c * 64;
#pragma unroll
    for (int u = 0; u < 8; ++u)
      *(bf16x8*)(dst + u * 8) = *(const bf16x8*)(src + u * 8);
  }
  __syncthreads();

  float tv[2][8]; int ti[2][8];
#pragma unroll
  for (int s = 0; s < 2; ++s)
#pragma unroll
    for (int j = 0; j < 8; ++j) { tv[s][j] = -FLT_MAX; ti[s][j] = 0; }

  const int splitStart = split * splitLen;
  const int splitEnd   = (splitStart + splitLen < N) ? (splitStart + splitLen) : N;
  const int iters      = (splitLen + NWAVES * 16 - 1) / (NWAVES * 16);
  float* myC0 = sC + (wave * 2 + 0) * 320;
  float* myC1 = sC + (wave * 2 + 1) * 320;

  struct AF { bf16x16 A0, A1; };

  for (int t = 0; t < iters; ++t) {
    const int keyBase = splitStart + t * (NWAVES * 16) + wave * 16;
    int krow = keyBase + row16;
    krow = (krow < N - 1) ? krow : (N - 1);
    const __bf16* kr = kn + (size_t)krow * D;

    floatx8 acc0 = {}, acc1 = {};
    bf16x16 B0[GRP], B1[GRP];
    AF Ab[2];

    auto loadB = [&](bf16x16* Bg, int g) {
#pragma unroll
      for (int j = 0; j < GRP; ++j)
        Bg[j] = *(const bf16x16*)(kr + (g * GRP + j) * 32 + hi * 16);
    };
    auto loadA = [&](AF& A, int kk) {
      const __bf16* a0 = sQ + row16 * LDSROW + kk * 32 + hi * 16;
      const __bf16* a1 = a0 + 16 * LDSROW;
      bf16x8 l0 = *(const bf16x8*)a0, h0 = *(const bf16x8*)(a0 + 8);
      bf16x8 l1 = *(const bf16x8*)a1, h1 = *(const bf16x8*)(a1 + 8);
      A.A0 = SHUF16(l0, h0);
      A.A1 = SHUF16(l1, h1);
    };

    // software pipeline: two B groups + one A step in flight
    loadB(B0, 0);
    loadB(B1, 1);
    loadA(Ab[0], 0);
#pragma unroll
    for (int kk = 0; kk < KSTEPS; ++kk) {
      const int cur = kk & 1;
      if (kk + 1 < KSTEPS) loadA(Ab[cur ^ 1], kk + 1);
      const bf16x16& Bf = (((kk >> 2) & 1) ? B1 : B0)[kk & 3];
      acc0 = __builtin_amdgcn_wmma_f32_16x16x32_bf16(
          false, Ab[cur].A0, false, Bf, (short)0, acc0, false, false);
      acc1 = __builtin_amdgcn_wmma_f32_16x16x32_bf16(
          false, Ab[cur].A1, false, Bf, (short)0, acc1, false, false);
      if ((kk & 3) == 3 && kk < 2 * GRP)
        loadB(((kk >> 2) & 1) ? B1 : B0, (kk >> 2) + 2);
    }

    // ---- candidate insertion: both subtiles, one barrier pair ----
#pragma unroll
    for (int v = 0; v < 8; ++v) {
      myC0[(v + 8 * hi) * 20 + row16] = acc0[v];
      myC1[(v + 8 * hi) * 20 + row16] = acc1[v];
    }
    __builtin_amdgcn_wave_barrier();
    const int kidx0 = keyBase + hi * 8;
    const float* rp0 = myC0 + row16 * 20 + hi * 8;
    const float* rp1 = myC1 + row16 * 20 + hi * 8;
#pragma unroll
    for (int j = 0; j < 8; ++j) {
      const int ci = kidx0 + j;
      float cv0 = rp0[j];
      float cv1 = rp1[j];
      if (ci >= splitEnd) { cv0 = -FLT_MAX; cv1 = -FLT_MAX; }
      TOP8_INSERT(cv0, ci, tv[0], ti[0]);
      TOP8_INSERT(cv1, ci, tv[1], ti[1]);
    }
    __builtin_amdgcn_wave_barrier();
  }

  __syncthreads();   // everyone done reading sQ before overlaying with lists
#pragma unroll
  for (int s = 0; s < 2; ++s)
#pragma unroll
    for (int j = 0; j < 8; ++j) {
      Cand* p = sLists + ((size_t)(wave * 32 + lane) * 2 + s) * 8 + j;
      p->v = tv[s][j]; p->i = ti[s][j];
    }
  __syncthreads();

  // ---- merge 128 candidates per row; wave 0 handles all 32 rows ----
  if (threadIdx.x < QTILE) {
    const int row = threadIdx.x;        // 0..31 within tile
    const int s = row >> 4, r16 = row & 15;
    float mv[8]; int mi[8];
#pragma unroll
    for (int j = 0; j < 8; ++j) { mv[j] = -FLT_MAX; mi[j] = 0; }
    for (int w = 0; w < NWAVES; ++w)
      for (int h = 0; h < 2; ++h) {
        const Cand* L = sLists + ((size_t)(w * 32 + (r16 + 16 * h)) * 2 + s) * 8;
#pragma unroll
        for (int j = 0; j < 8; ++j) {
          float cv = L[j].v; int ci = L[j].i;
          TOP8_INSERT(cv, ci, mv, mi);
        }
      }
    Cand* o = cand + ((size_t)(qtile * gridDim.y + split) * QTILE + row) * 8;
#pragma unroll
    for (int j = 0; j < 8; ++j) { o[j].v = mv[j]; o[j].i = mi[j]; }
  }
}

// ------------- merge splits, softmax(8), weighted gather of values -------------
__global__ __launch_bounds__(256) void ltm_finalize(
    const Cand* __restrict__ cand, const float* __restrict__ values,
    float* __restrict__ out, int nsplits, int B) {
  const int lane = threadIdx.x & 31;
  const int wave = threadIdx.x >> 5;
  const int row  = blockIdx.x * NWAVES + wave;
  if (row >= B) return;
  const int qtile = row >> 5, r32 = row & 31;

  float tv[8]; int ti[8];
#pragma unroll
  for (int j = 0; j < 8; ++j) { tv[j] = -FLT_MAX; ti[j] = 0; }
  for (int s = 0; s < nsplits; ++s) {
    const Cand* L = cand + ((size_t)(qtile * nsplits + s) * QTILE + r32) * 8;
#pragma unroll
    for (int j = 0; j < 8; ++j) {
      float cv = L[j].v; int ci = L[j].i;
      TOP8_INSERT(cv, ci, tv, ti);
    }
  }
  float w[8], sum = 0.f;
#pragma unroll
  for (int j = 0; j < 8; ++j) { w[j] = __expf(tv[j] - tv[0]); sum += w[j]; }
  const float inv = 1.0f / sum;

  float accv[D / 32];
#pragma unroll
  for (int k = 0; k < D / 32; ++k) accv[k] = 0.f;
#pragma unroll
  for (int j = 0; j < 8; ++j) {
    const float* vr = values + (size_t)ti[j] * D;
    const float wj = w[j] * inv;
#pragma unroll
    for (int k = 0; k < D / 32; ++k) accv[k] += wj * vr[lane + 32 * k];
  }
  float* orow = out + (size_t)row * D;
#pragma unroll
  for (int k = 0; k < D / 32; ++k) orow[lane + 32 * k] = accv[k];
}

static inline size_t align256(size_t x) { return (x + 255) & ~(size_t)255; }

extern "C" void kernel_launch(void* const* d_in, const int* in_sizes, int n_in,
                              void* d_out, int out_size, void* d_ws, size_t ws_size,
                              hipStream_t stream) {
  const float* q = (const float*)d_in[0];
  const float* k = (const float*)d_in[1];
  const float* v = (const float*)d_in[2];
  // d_in[3] = top_k, fixed at 8 by the reference problem
  const int B = in_sizes[0] / D;   // 2048
  const int N = in_sizes[1] / D;   // 100000
  float* out = (float*)d_out;

  char* ws = (char*)d_ws;
  __bf16* qn = (__bf16*)ws;
  size_t off = align256((size_t)B * D * sizeof(__bf16));
  __bf16* kn = (__bf16*)(ws + off);
  off += align256((size_t)N * D * sizeof(__bf16));
  Cand* cand = (Cand*)(ws + off);

  const int splitLen = (N + NSPLITS - 1) / NSPLITS;

  ltm_normalize_bf16<<<(B + NWAVES - 1) / NWAVES, 256, 0, stream>>>(q, qn, B);
  ltm_normalize_bf16<<<(N + NWAVES - 1) / NWAVES, 256, 0, stream>>>(k, kn, N);

  dim3 grid(B / QTILE, NSPLITS);
  ltm_topk_partial<<<grid, 256, 0, stream>>>(qn, kn, cand, N, splitLen);

  ltm_finalize<<<(B + NWAVES - 1) / NWAVES, 256, 0, stream>>>(cand, v, out, NSPLITS, B);
}
